// FSMNSeleNetV3_39599598469381
// MI455X (gfx1250) — matrix operand order repaired and checked by reading, not compiled, not run on hardware
//
#include <hip/hip_runtime.h>

// FSMN SeleNet V3 for gfx1250 (MI455X): wave32, WMMA f32_16x16x32_f16, fused units,
// async global->LDS staging (GLOBAL_LOAD_ASYNC_TO_LDS_B128 + s_wait_asynccnt).
//
// Pipeline (7 launches on `stream`):
//   unit<0>: x(f32) -> hA      expand 120->128 relu, shrink ->64, fused fsmn conv (no residual)
//   unit<1>: hA->hB, hB->hA, hA->hB, hB->hA   (fused conv + residual, x4 ping-pong)
//   unit<2>: hA -> hC          relu(h@We2+be2) f16 [128,2048,128]
//   decode : channel-max over C=4 + 128->5 GEMV + bias -> d_out f32 [32,2048,5]
// Workspace: hA @0 (32MB), hB @32MB (32MB), hC @32MB (64MB; hB dead) => needs 96MB.

typedef _Float16 v16h __attribute__((ext_vector_type(16)));
typedef _Float16 v8h  __attribute__((ext_vector_type(8)));
typedef float    v8f  __attribute__((ext_vector_type(8)));

#define TM   64     // interior tokens per block
#define LDK  136    // padded K stride (halfs); 272B keeps A-frag row reads conflict-free
#define LDO  72     // padded stride for the o tile (halfs)
#define TSEQ 2048
#define NSEQ 128    // B*C

// A fragment (16x32 f16, MxK): lane L -> M = L&15; halfs 0..7: K=kb+0..7, halfs 8..15:
// K=kb+16..23, kb = ks*32 + (L>=16 ? 8 : 0).
__device__ __forceinline__ v16h load_frag_a(const _Float16* __restrict__ base, int lane, int ks) {
    int m  = lane & 15;
    int kb = (ks << 5) + ((lane >> 4) << 3);
    const _Float16* p = base + m * LDK + kb;
    v8h lo = *(const v8h*)p;
    v8h hi = *(const v8h*)(p + 16);
    return __builtin_shufflevector(lo, hi, 0,1,2,3,4,5,6,7,8,9,10,11,12,13,14,15);
}

// B fragment (32x16 f16, KxN), weights [N][K]-major in LDS: lane L -> N = L&15,
// halfs 0..7: K=kb+0..7, halfs 8..15: K=kb+16..23, kb = ks*32 + (L>=16 ? 8 : 0).
__device__ __forceinline__ v16h load_frag_b(const _Float16* __restrict__ base, int lane, int ks) {
    int n  = lane & 15;
    int kb = (ks << 5) + ((lane >> 4) << 3);
    const _Float16* p = base + n * LDK + kb;
    v8h lo = *(const v8h*)p;
    v8h hi = *(const v8h*)(p + 16);
    return __builtin_shufflevector(lo, hi, 0,1,2,3,4,5,6,7,8,9,10,11,12,13,14,15);
}

// Async global->LDS copy of 16 bytes; tracked with ASYNCcnt.
__device__ __forceinline__ void async_b128(unsigned lds_addr, int goff_bytes, const void* sbase) {
    asm volatile("global_load_async_to_lds_b128 %0, %1, %2 offset:0"
                 :: "v"(lds_addr), "v"(goff_bytes), "s"(sbase) : "memory");
}
__device__ __forceinline__ void wait_async0() {
    asm volatile("s_wait_asynccnt 0x0" ::: "memory");
}

// MODE 0: A from x (f32, pad 120->128), halo rows, fused fsmn conv, no residual
// MODE 1: A from hin (f16, K=64, async),  halo rows, fused fsmn conv + residual
// MODE 2: A from hin (f16, K=64, async),  expand only, write relu(E) [S,T,128]
template <int MODE>
__global__ __launch_bounds__(256) void fsmn_unit_kernel(
    const float*    __restrict__ x,     // MODE 0
    const _Float16* __restrict__ hin,   // MODE 1/2
    const float*    __restrict__ Wexp,  // [Kreal, 128]
    const float*    __restrict__ bexp,  // [128]
    const float*    __restrict__ Wshr,  // [128, 64]  (MODE 0/1)
    const float*    __restrict__ wl,    // [10, 64]   (MODE 0/1)
    const float*    __restrict__ wrw,   // [1, 64]    (MODE 0/1)
    _Float16*       __restrict__ outp)  // MODE 0/1: [S,T,64]; MODE 2: [S,T,128]
{
    constexpr int HALO  = (MODE == 2) ? 0 : 12;   // rows [t0-12, t0+68); conv needs [-9,+1]
    constexpr int MSUB  = (MODE == 2) ? 4 : 5;
    constexpr int MROWS = MSUB * 16;              // 64 or 80
    constexpr int Kexp  = (MODE == 0) ? 128 : 64;
    constexpr int Kreal = (MODE == 0) ? 120 : 64;
    constexpr int KS    = Kexp >> 5;

    __shared__ __attribute__((aligned(16))) _Float16 sA[MROWS * LDK];  // A tile -> E tile
    __shared__ __attribute__((aligned(16))) _Float16 sW[128 * LDK];    // We -> Ws -> o tile
    __shared__ float sB[128];
    __shared__ float sWl[(MODE == 2) ? 1 : 640];
    __shared__ float sWr[(MODE == 2) ? 1 : 64];

    const int tid  = threadIdx.x;
    const int lane = tid & 31;
    const int w    = tid >> 5;          // wave 0..7
    const int t0   = blockIdx.x * TM;
    const int s    = blockIdx.y;

    // ---- stage A tile ----
    if constexpr (MODE == 0) {
        const int b = s >> 2, c = s & 3;
        for (int idx = tid; idx < MROWS * 128; idx += 256) {
            int r = idx >> 7, k = idx & 127;
            int tr = t0 - HALO + r;
            tr = tr < 0 ? 0 : (tr > TSEQ - 1 ? TSEQ - 1 : tr);  // conv masks OOB taps
            float v = (k < 120) ? x[(((size_t)(b * TSEQ + tr)) * 4 + c) * 120 + k] : 0.f;
            sA[r * LDK + k] = (_Float16)v;
        }
    } else {
        const _Float16* seqBase = hin + (size_t)s * TSEQ * 64;
        for (int chunk = tid; chunk < (MROWS * 64) / 8; chunk += 256) {
            int r = chunk >> 3, cc = (chunk & 7) << 3;   // 8 halfs = 16B per chunk
            int tr = t0 - HALO + r;
            tr = tr < 0 ? 0 : (tr > TSEQ - 1 ? TSEQ - 1 : tr);
            unsigned lds = (unsigned)(size_t)&sA[r * LDK + cc];
            async_b128(lds, (tr * 64 + cc) * 2, (const void*)seqBase);
        }
    }
    // ---- stage expand weights [N][K]-major, bias, conv weights ----
    for (int idx = tid; idx < 128 * Kexp; idx += 256) {
        int k = idx >> 7, n = idx & 127;
        float v = (k < Kreal) ? Wexp[k * 128 + n] : 0.f;
        sW[n * LDK + k] = (_Float16)v;
    }
    if (tid < 128) sB[tid] = bexp[tid];
    if constexpr (MODE != 2) {
        for (int idx = tid; idx < 640; idx += 256) sWl[idx] = wl[idx];
        if (tid < 64) sWr[tid] = wrw[tid];
    }
    if constexpr (MODE != 0) wait_async0();
    __syncthreads();

    // ---- phase 1: E = relu(A @ We + b); wave w owns N-strip w, mi = 0..MSUB-1 ----
    v8f acc[MSUB];
    #pragma unroll
    for (int mi = 0; mi < MSUB; ++mi) {
        v8f c = {};
        #pragma unroll
        for (int ks = 0; ks < KS; ++ks) {
            v16h a = load_frag_a(sA + mi * 16 * LDK, lane, ks);
            v16h b = load_frag_b(sW + w * 16 * LDK, lane, ks);
            c = __builtin_amdgcn_wmma_f32_16x16x32_f16(false, a, false, b, (short)0, c, false, false);
        }
        acc[mi] = c;
    }

    const int nCol = (w << 4) + (lane & 15);
    const float bias = sB[nCol];
    const int mOff = (lane >> 4) << 3;   // upper half-wave holds rows +8

    if constexpr (MODE == 2) {
        #pragma unroll
        for (int mi = 0; mi < MSUB; ++mi)
            #pragma unroll
            for (int r = 0; r < 8; ++r) {
                float e = acc[mi][r] + bias; e = e > 0.f ? e : 0.f;
                outp[((size_t)s * TSEQ + t0 + mi * 16 + mOff + r) * 128 + nCol] = (_Float16)e;
            }
        return;
    } else {
        __syncthreads();  // all waves done reading sA (A) and sW (We)
        #pragma unroll
        for (int mi = 0; mi < MSUB; ++mi)
            #pragma unroll
            for (int r = 0; r < 8; ++r) {
                float e = acc[mi][r] + bias; e = e > 0.f ? e : 0.f;
                sA[(mi * 16 + mOff + r) * LDK + nCol] = (_Float16)e;   // sA now holds E
            }
        // reload sW with Ws [128,64] -> [N=64][K=128]
        for (int idx = tid; idx < 128 * 64; idx += 256) {
            int k = idx >> 6, n = idx & 63;
            sW[n * LDK + k] = (_Float16)Wshr[k * 64 + n];
        }
        __syncthreads();

        // ---- phase 2: O = E @ Ws (M=80, N=64, K=128): 20 tiles; wave w -> {w, w+8, w+16} ----
        v8f oacc[3];
        const int nt = (w < 4) ? 3 : 2;
        for (int i = 0; i < nt; ++i) {
            int tIdx = w + (i << 3);
            int mi = tIdx % 5, ni = tIdx / 5;
            v8f c = {};
            #pragma unroll
            for (int ks = 0; ks < 4; ++ks) {
                v16h a = load_frag_a(sA + mi * 16 * LDK, lane, ks);
                v16h b = load_frag_b(sW + ni * 16 * LDK, lane, ks);
                c = __builtin_amdgcn_wmma_f32_16x16x32_f16(false, a, false, b, (short)0, c, false, false);
            }
            oacc[i] = c;
        }
        __syncthreads();  // all waves done reading E (sA) and Ws (sW)

        // ---- stage o tile [MROWS][64] into sW region (stride LDO) ----
        _Float16* sO = (_Float16*)sW;
        for (int i = 0; i < nt; ++i) {
            int tIdx = w + (i << 3);
            int mi = tIdx % 5, ni = tIdx / 5;
            int n2 = (ni << 4) + (lane & 15);
            int mbase = mi * 16 + mOff;
            #pragma unroll
            for (int r = 0; r < 8; ++r)
                sO[(mbase + r) * LDO + n2] = (_Float16)oacc[i][r];
        }
        __syncthreads();

        // ---- fused FSMN conv (+ residual) over interior rows; per-sequence zero padding ----
        const int d = tid & 63;
        for (int i = tid >> 6; i < TM; i += 4) {
            int t = t0 + i;
            float a = (float)sO[(i + HALO) * LDO + d];          // o[t]
            #pragma unroll
            for (int j = 0; j < 10; ++j) {
                int tt = t - 9 + j;
                if ((unsigned)tt < (unsigned)TSEQ)
                    a += sWl[j * 64 + d] * (float)sO[(i + HALO - 9 + j) * LDO + d];
            }
            if (t + 1 < TSEQ)
                a += sWr[d] * (float)sO[(i + HALO + 1) * LDO + d];
            if constexpr (MODE == 1)
                a += (float)hin[((size_t)s * TSEQ + t) * 64 + d];  // residual
            outp[((size_t)s * TSEQ + t) * 64 + d] = (_Float16)a;
        }
    }
}

// channel max over C=4 then 128->5 GEMV + bias
__global__ __launch_bounds__(128) void decode_kernel(
    const _Float16* __restrict__ e,   // [128, T, 128], seq = b*4+c
    const float*    __restrict__ Wd,  // [128,5]
    const float*    __restrict__ bd,  // [5]
    float*          __restrict__ out) // [32,T,5]
{
    __shared__ float sm[128];
    int tok = blockIdx.x;             // 0..65535
    int b = tok >> 11, t = tok & 2047;
    int d = threadIdx.x;
    size_t base = (((size_t)(b * 4)) * TSEQ + t) * 128 + d;
    const size_t cstep = (size_t)TSEQ * 128;
    float m = (float)e[base];
    m = fmaxf(m, (float)e[base + cstep]);
    m = fmaxf(m, (float)e[base + 2 * cstep]);
    m = fmaxf(m, (float)e[base + 3 * cstep]);
    sm[d] = m;
    __syncthreads();
    if (d < 5) {
        float acc = bd[d];
        #pragma unroll 4
        for (int k = 0; k < 128; ++k) acc += sm[k] * Wd[k * 5 + d];
        out[(size_t)tok * 5 + d] = acc;
    }
}

extern "C" void kernel_launch(void* const* d_in, const int* in_sizes, int n_in,
                              void* d_out, int out_size, void* d_ws, size_t ws_size,
                              hipStream_t stream) {
    const float* x   = (const float*)d_in[0];
    const float* We0 = (const float*)d_in[1];
    const float* be0 = (const float*)d_in[2];
    const float* Ws0 = (const float*)d_in[3];
    const float* wl0 = (const float*)d_in[4];
    const float* wr0 = (const float*)d_in[5];
    const float* We  = (const float*)d_in[6];   // [4,64,128]
    const float* be  = (const float*)d_in[7];   // [4,128]
    const float* Ws  = (const float*)d_in[8];   // [4,128,64]
    const float* wl  = (const float*)d_in[9];   // [4,10,64]
    const float* wr  = (const float*)d_in[10];  // [4,1,64]
    const float* We2 = (const float*)d_in[11];
    const float* be2 = (const float*)d_in[12];
    const float* Wd  = (const float*)d_in[13];
    const float* bd  = (const float*)d_in[14];
    float* out = (float*)d_out;

    char* ws = (char*)d_ws;
    _Float16* hA = (_Float16*)ws;                        // 32MB [128,2048,64]
    _Float16* hB = (_Float16*)(ws + ((size_t)32 << 20)); // 32MB [128,2048,64]
    _Float16* hC = (_Float16*)(ws + ((size_t)32 << 20)); // 64MB [128,2048,128] (hB dead by then)

    dim3 ggrid(TSEQ / TM, NSEQ);  // (32, 128)

    // unit 0 (fused conv, no residual): x -> hA
    fsmn_unit_kernel<0><<<ggrid, 256, 0, stream>>>(x, nullptr, We0, be0, Ws0, wl0, wr0, hA);
    // units 1..4 (fused conv + residual), ping-pong hA <-> hB
    _Float16* bufs[2] = { hA, hB };
    for (int l = 0; l < 4; ++l) {
        _Float16* src = bufs[l & 1];
        _Float16* dst = bufs[(l & 1) ^ 1];
        fsmn_unit_kernel<1><<<ggrid, 256, 0, stream>>>(nullptr, src, We + l * 64 * 128,
                                                       be + l * 128, Ws + l * 128 * 64,
                                                       wl + l * 10 * 64, wr + l * 64, dst);
    }
    // after 4 units result is back in hA; final expand + relu -> hC
    fsmn_unit_kernel<2><<<ggrid, 256, 0, stream>>>(nullptr, hA, We2, be2, nullptr,
                                                   nullptr, nullptr, hC);
    // channel max + decode
    decode_kernel<<<32 * TSEQ, 128, 0, stream>>>(hC, Wd, bd, out);
}
